// GCNExtractor_89163521065534
// MI455X (gfx1250) — compile-verified, matrix operand order checked by reading, not run on hardware
//
#include <hip/hip_runtime.h>

#define B_  4
#define C_  256
#define H_  512
#define W_  256
#define K_  64
#define CR  64
#define KNN 8

typedef __attribute__((ext_vector_type(16))) __bf16 v16bf;
typedef __attribute__((ext_vector_type(8)))  float  v8f;

// ---------------------------------------------------------------------------
// Kernel 1: width-mean + grouped 1x1 reduce + support GEMM (Xr @ W_gcn) fused.
// One block per (b,h). 8 waves stream 256 channels x 256 floats (coalesced
// float4), wave-reduce, then 64 threads do the reduce + 64x64 GEMM row.
// ---------------------------------------------------------------------------
__global__ void k_mean_reduce_support(const float* __restrict__ x,
                                      const float* __restrict__ Wred,
                                      const float* __restrict__ Wgcn,
                                      float* __restrict__ S)
{
    __shared__ float chanMean[C_];
    __shared__ float xr[CR];
    const int bh  = blockIdx.x;
    const int b   = bh / H_, h = bh % H_;
    const int tid = threadIdx.x, wv = tid >> 5, lane = tid & 31;

    for (int c = wv; c < C_; c += 8) {
        const float4* xp = (const float4*)(x + ((size_t)(b * C_ + c) * H_ + h) * W_);
        float4 f0 = xp[lane];
        float4 f1 = xp[lane + 32];
        float s = f0.x + f0.y + f0.z + f0.w + f1.x + f1.y + f1.z + f1.w;
        for (int off = 16; off > 0; off >>= 1) s += __shfl_down(s, off);
        if (lane == 0) chanMean[c] = s * (1.0f / (float)W_);
    }
    __syncthreads();
    if (tid < CR) {
        float acc = 0.0f;
        for (int g = 0; g < 4; ++g) acc += chanMean[tid * 4 + g] * Wred[tid * 4 + g];
        xr[tid] = acc;
    }
    __syncthreads();
    if (tid < CR) {
        float acc = 0.0f;
        for (int c = 0; c < CR; ++c) acc += xr[c] * Wgcn[c * CR + tid];
        S[((size_t)b * H_ + h) * CR + tid] = acc;
    }
}

// ---------------------------------------------------------------------------
// Kernel 2: row-normalize Uk and convert to bf16 for WMMA.
// One wave per row (K=64 -> 2 elements/lane), full-wave xor reduction.
// ---------------------------------------------------------------------------
__global__ void k_normalize_bf16(const float* __restrict__ Uk, __bf16* __restrict__ ubf)
{
    const int row  = blockIdx.x;      // b*H + h
    const int lane = threadIdx.x;     // 0..31
    const float* up = Uk + (size_t)row * K_;
    float v0 = up[lane], v1 = up[lane + 32];
    float ss = v0 * v0 + v1 * v1;
    for (int off = 16; off > 0; off >>= 1) ss += __shfl_xor(ss, off);
    float n = sqrtf(ss);
    if (n < 1e-12f) n = 1e-12f;
    const float inv = 1.0f / n;
    ubf[(size_t)row * K_ + lane]      = (__bf16)(v0 * inv);
    ubf[(size_t)row * K_ + lane + 32] = (__bf16)(v1 * inv);
}

// ---------------------------------------------------------------------------
// Kernel 3: sim = U U^T via v_wmma_f32_16x16x32_bf16, then per-row top-8.
// One block per (b, 16-row tile). 4 waves x 8 column tiles; each 16x16 tile
// is two chained WMMAs over K=64. Results staged in LDS (padded stride),
// then 16 threads run insertion top-8 (tie-break: lower index first).
// ---------------------------------------------------------------------------
#define SIM_LD 514
__global__ void k_sim_topk(const __bf16* __restrict__ ubf,
                           float* __restrict__ vals,
                           int*   __restrict__ idx,
                           float* __restrict__ dinv)
{
    __shared__ float sim[16 * SIM_LD];
    const int b       = blockIdx.x >> 5;
    const int rowBase = (blockIdx.x & 31) * 16;
    const int tid = threadIdx.x, wv = tid >> 5, lane = tid & 31;
    const int m = lane & 15, hi = lane >> 4, kb = hi * 8;

    const __bf16* Ub = ubf + (size_t)b * H_ * K_;

    // A fragments (rows of this tile), ISA 16-bit A layout: lane m (+16 for
    // K-half), VGPRs 0-3 hold K = kb+0..7, VGPRs 4-7 hold K = 16+kb+0..7.
    const __bf16* ar = Ub + (size_t)(rowBase + m) * K_;
    v16bf a0, a1;
    for (int e = 0; e < 8; ++e) {
        a0[e]     = ar[kb + e];
        a0[e + 8] = ar[16 + kb + e];
        a1[e]     = ar[32 + kb + e];
        a1[e + 8] = ar[48 + kb + e];
    }

    for (int ct = wv; ct < 32; ct += 4) {
        const __bf16* br = Ub + (size_t)(ct * 16 + m) * K_;  // B col n = lane&15
        v16bf b0, b1;
        for (int e = 0; e < 8; ++e) {
            b0[e]     = br[kb + e];
            b0[e + 8] = br[16 + kb + e];
            b1[e]     = br[32 + kb + e];
            b1[e + 8] = br[48 + kb + e];
        }
        v8f c = {};
        c = __builtin_amdgcn_wmma_f32_16x16x32_bf16(false, a0, false, b0,
                                                    (short)0, c, false, false);
        c = __builtin_amdgcn_wmma_f32_16x16x32_bf16(false, a1, false, b1,
                                                    (short)0, c, false, false);
        // C/D layout: VGPR r -> row r + 8*hi, lane&15 -> col
        for (int r = 0; r < 8; ++r)
            sim[(r + 8 * hi) * SIM_LD + ct * 16 + m] = c[r];
    }
    __syncthreads();

    if (tid < 16) {
        float tv[KNN];
        int   ti[KNN];
        for (int j = 0; j < KNN; ++j) { tv[j] = -1e30f; ti[j] = 0; }
        const float* srow = sim + tid * SIM_LD;
        for (int g = 0; g < H_; ++g) {
            float v = srow[g];
            if (v > tv[KNN - 1]) {
                int j = KNN - 1;
                while (j > 0 && v > tv[j - 1]) {
                    tv[j] = tv[j - 1]; ti[j] = ti[j - 1]; --j;
                }
                tv[j] = v; ti[j] = g;
            }
        }
        float deg = 0.0f;
        for (int j = 0; j < KNN; ++j) deg += tv[j];
        const int row = rowBase + tid;
        const size_t o = ((size_t)b * H_ + row) * KNN;
        for (int j = 0; j < KNN; ++j) { vals[o + j] = tv[j]; idx[o + j] = ti[j]; }
        dinv[(size_t)b * H_ + row] = rsqrtf(deg);
    }
}

// ---------------------------------------------------------------------------
// Kernel 4: neighbor aggregation + bias + ReLU + grouped expand + broadcast
// store over W. One block per (b,h); stores are 64-thread-contiguous float4
// bursts per channel (fully coalesced; this kernel is write-bound by design).
// ---------------------------------------------------------------------------
__global__ void k_aggregate_expand(const float* __restrict__ S,
                                   const float* __restrict__ vals,
                                   const int*   __restrict__ idx,
                                   const float* __restrict__ dinv,
                                   const float* __restrict__ bg,
                                   const float* __restrict__ Wexp,
                                   float* __restrict__ out)
{
    __shared__ float wk[KNN];
    __shared__ int   ik[KNN];
    __shared__ float o64[CR];
    __shared__ float ex[C_];
    const int bh  = blockIdx.x;
    const int b   = bh / H_, h = bh % H_;
    const int tid = threadIdx.x;

    if (tid < KNN) {
        const size_t o = (size_t)bh * KNN + tid;
        const int j = idx[o];
        ik[tid] = j;
        wk[tid] = vals[o] * dinv[bh] * dinv[(size_t)b * H_ + j];
    }
    __syncthreads();
    if (tid < CR) {
        float acc = bg[tid];
        for (int k = 0; k < KNN; ++k)
            acc += wk[k] * S[((size_t)b * H_ + ik[k]) * CR + tid];
        o64[tid] = acc > 0.0f ? acc : 0.0f;
    }
    __syncthreads();
    {
        const int g = tid >> 2, j = tid & 3;
        ex[tid] = o64[g] * Wexp[g * 4 + j];
    }
    __syncthreads();

    const int cg = tid >> 6;   // 0..3 : channel within group of 4
    const int wq = tid & 63;   // 0..63: float4 index along W
    for (int cb = 0; cb < C_; cb += 4) {
        const int c = cb + cg;
        const float v = ex[c];
        float4 vv = make_float4(v, v, v, v);
        float4* p = (float4*)(out + ((size_t)(b * C_ + c) * H_ + h) * W_);
        p[wq] = vv;
    }
}

// ---------------------------------------------------------------------------
extern "C" void kernel_launch(void* const* d_in, const int* in_sizes, int n_in,
                              void* d_out, int out_size, void* d_ws, size_t ws_size,
                              hipStream_t stream)
{
    (void)in_sizes; (void)n_in; (void)out_size; (void)ws_size;
    const float* x    = (const float*)d_in[0];
    const float* Uk   = (const float*)d_in[1];
    const float* Wred = (const float*)d_in[2];
    const float* Wgcn = (const float*)d_in[3];
    const float* bg   = (const float*)d_in[4];
    const float* Wexp = (const float*)d_in[5];
    float* out = (float*)d_out;

    char* ws = (char*)d_ws;
    float*  S    = (float*)(ws);              // 4*512*64*4  = 524288 B
    __bf16* ubf  = (__bf16*)(ws + 524288);    // 4*512*64*2  = 262144 B
    float*  vals = (float*)(ws + 786432);     // 4*512*8*4   =  65536 B
    int*    idxb = (int*)  (ws + 851968);     // 4*512*8*4   =  65536 B
    float*  dinv = (float*)(ws + 917504);     // 4*512*4     =   8192 B

    k_mean_reduce_support<<<B_ * H_, 256, 0, stream>>>(x, Wred, Wgcn, S);
    k_normalize_bf16<<<B_ * H_, 32, 0, stream>>>(Uk, ubf);
    k_sim_topk<<<B_ * (H_ / 16), 128, 0, stream>>>(ubf, vals, idxb, dinv);
    k_aggregate_expand<<<B_ * H_, 256, 0, stream>>>(S, vals, idxb, dinv, bg, Wexp, out);
}
